// GCNESAModel_20246475833773
// MI455X (gfx1250) — matrix operand order repaired and checked by reading, not compiled, not run on hardware
//
#include <hip/hip_runtime.h>
#include <math.h>

// ---------------- model constants ----------------
constexpr int G     = 4;
constexpr int L     = 128;
constexpr int APB   = 8;
constexpr int H     = 128;
constexpr int HEADS = 4;
constexpr int KNN   = 16;
constexpr int NPG   = L * APB;      // 1024 atoms per graph
constexpr int NA    = G * NPG;      // 4096 atoms
constexpr int NB    = G * L;        // 512 blocks
constexpr int HD    = H / HEADS;    // 32
constexpr int E     = L * KNN;      // 2048 edge slots per graph
constexpr int EFPAD = 288;          // 257 -> padded to multiple of 32
constexpr float EPS   = 1e-5f;
constexpr float SCALE = 0.17677669529663687f;   // 1/sqrt(32)

typedef __attribute__((ext_vector_type(16))) _Float16 v16h;
typedef __attribute__((ext_vector_type(8)))  _Float16 v8h;
typedef __attribute__((ext_vector_type(8)))  float    v8f;
typedef __attribute__((ext_vector_type(4)))  float    v4f;

#define ACT_NONE 0
#define ACT_RELU 1

// =====================================================================
// Weight packer: Wt[n*Kpad + k] = f16(W[k*N + n]), zero pad k >= Kd.
// =====================================================================
__global__ void pack_w_kernel(const float* __restrict__ W, _Float16* __restrict__ Wt,
                              int Kd, int Kpad, int N)
{
    int t = blockIdx.x * blockDim.x + threadIdx.x;
    if (t >= N * Kpad) return;
    int n = t / Kpad, k = t % Kpad;
    Wt[t] = (_Float16)((k < Kd) ? W[(size_t)k * N + n] : 0.f);
}

// =====================================================================
// Tiled WMMA GEMM:  C[M,N] = act(A[M,(lda)] @ W[Kd,N] + bias) (+res)
// Kd % 32 == 0, N % 64 == 0.  One wave computes a 16x64 tile of C
// (4 accumulators): A loaded once per K-chunk, 4 WMMAs issued per chunk.
// A: f32 row-major (stride lda); Wt: pre-packed f16 [N][Kd].
// =====================================================================
__global__ void gemm_wmma_kernel(const float* __restrict__ A, int lda,
                                 const _Float16* __restrict__ Wt,
                                 const float* __restrict__ bias,
                                 const float* __restrict__ res,
                                 float* __restrict__ C,
                                 int M, int Kd, int N, int act)
{
    int wave = (int)((blockIdx.x * blockDim.x + threadIdx.x) >> 5);
    int lane = (int)(threadIdx.x & 31);
    int ngrp = N >> 6;                        // 64-wide column groups
    int mt   = wave / ngrp;
    int ng   = wave % ngrp;
    if (mt * 16 >= M) return;                 // wave-uniform exit

    int half = lane >> 4;
    int ln   = lane & 15;
    const float*    arow = A  + (size_t)(mt * 16 + ln) * lda + half * 16;
    const _Float16* bcol = Wt + (size_t)(ng * 64 + ln) * Kd  + half * 16;
    size_t bstep = (size_t)16 * Kd;           // next 16-column tile

    v8f acc0 = {}, acc1 = {}, acc2 = {}, acc3 = {};
#pragma unroll 2
    for (int k0 = 0; k0 < Kd; k0 += 32) {
        v4f a0 = *(const v4f*)(arow + k0);
        v4f a1 = *(const v4f*)(arow + k0 + 4);
        v4f a2 = *(const v4f*)(arow + k0 + 8);
        v4f a3 = *(const v4f*)(arow + k0 + 12);
        v16h a;
#pragma unroll
        for (int i = 0; i < 4; ++i) {
            a[i]      = (_Float16)a0[i];
            a[4 + i]  = (_Float16)a1[i];
            a[8 + i]  = (_Float16)a2[i];
            a[12 + i] = (_Float16)a3[i];
        }
        v16h b;
        v8h p, q;
        // tile 0
        p = *(const v8h*)(bcol + k0);  q = *(const v8h*)(bcol + k0 + 8);
#pragma unroll
        for (int i = 0; i < 8; ++i) { b[i] = p[i]; b[8 + i] = q[i]; }
        acc0 = __builtin_amdgcn_wmma_f32_16x16x32_f16(false, a, false, b, (short)0, acc0, false, false);
        // tile 1
        p = *(const v8h*)(bcol + bstep + k0);  q = *(const v8h*)(bcol + bstep + k0 + 8);
#pragma unroll
        for (int i = 0; i < 8; ++i) { b[i] = p[i]; b[8 + i] = q[i]; }
        acc1 = __builtin_amdgcn_wmma_f32_16x16x32_f16(false, a, false, b, (short)0, acc1, false, false);
        // tile 2
        p = *(const v8h*)(bcol + 2 * bstep + k0);  q = *(const v8h*)(bcol + 2 * bstep + k0 + 8);
#pragma unroll
        for (int i = 0; i < 8; ++i) { b[i] = p[i]; b[8 + i] = q[i]; }
        acc2 = __builtin_amdgcn_wmma_f32_16x16x32_f16(false, a, false, b, (short)0, acc2, false, false);
        // tile 3
        p = *(const v8h*)(bcol + 3 * bstep + k0);  q = *(const v8h*)(bcol + 3 * bstep + k0 + 8);
#pragma unroll
        for (int i = 0; i < 8; ++i) { b[i] = p[i]; b[8 + i] = q[i]; }
        acc3 = __builtin_amdgcn_wmma_f32_16x16x32_f16(false, a, false, b, (short)0, acc3, false, false);
    }
    v8f accs[4] = {acc0, acc1, acc2, acc3};
#pragma unroll
    for (int t = 0; t < 4; ++t) {
#pragma unroll
        for (int j = 0; j < 8; ++j) {
            int r = mt * 16 + half * 8 + j;
            int c = ng * 64 + t * 16 + ln;
            float v = accs[t][j] + (bias ? bias[c] : 0.f);
            if (act == ACT_RELU) v = fmaxf(v, 0.f);
            if (res) v += res[(size_t)r * N + c];
            C[(size_t)r * N + c] = v;
        }
    }
}

// =====================================================================
// Embedding
// =====================================================================
__global__ void embed_kernel(const int* __restrict__ A, const int* __restrict__ AP,
                             const int* __restrict__ B,
                             const float* __restrict__ ae, const float* __restrict__ pe,
                             const float* __restrict__ be, float* __restrict__ x)
{
    int t = blockIdx.x * blockDim.x + threadIdx.x;
    if (t >= NA * H) return;
    int i = t / H, f = t % H;
    x[t] = ae[A[i] * H + f] + pe[AP[i] * H + f] + be[B[i / APB] * H + f];
}

// =====================================================================
// k-NN + radius mask (register-resident top-16 insertion sort)
// =====================================================================
__global__ void knn_kernel(const float* __restrict__ coords,
                           int* __restrict__ idx, float* __restrict__ mask,
                           float* __restrict__ deg, int n, float r2)
{
    int g = blockIdx.y;
    int i = blockIdx.x * blockDim.x + threadIdx.x;
    if (i >= n) return;
    const float* cg = coords + (size_t)g * n * 3;
    float xi = cg[i * 3 + 0], yi = cg[i * 3 + 1], zi = cg[i * 3 + 2];

    float bd[KNN]; int bi[KNN];
#pragma unroll
    for (int t = 0; t < KNN; ++t) { bd[t] = 3.0e38f; bi[t] = 0; }

    for (int j = 0; j < n; ++j) {
        if (j == i) continue;
        float dx = cg[j * 3 + 0] - xi;
        float dy = cg[j * 3 + 1] - yi;
        float dz = cg[j * 3 + 2] - zi;
        float d2 = dx * dx + dy * dy + dz * dz;
        if (d2 < bd[KNN - 1]) {
            bd[KNN - 1] = d2; bi[KNN - 1] = j;
#pragma unroll
            for (int t = KNN - 1; t > 0; --t) {
                if (bd[t] < bd[t - 1]) {
                    float td = bd[t]; bd[t] = bd[t - 1]; bd[t - 1] = td;
                    int   ti = bi[t]; bi[t] = bi[t - 1]; bi[t - 1] = ti;
                }
            }
        }
    }
    float dsum = 1.0f;
#pragma unroll
    for (int t = 0; t < KNN; ++t) {
        float m = (bd[t] <= r2) ? 1.0f : 0.0f;
        idx [((size_t)g * n + i) * KNN + t] = bi[t];
        mask[((size_t)g * n + i) * KNN + t] = m;
        dsum += m;
    }
    deg[(size_t)g * n + i] = dsum;
}

// =====================================================================
// GCN aggregation
// =====================================================================
__global__ void gcn_agg_kernel(const float* __restrict__ xw,
                               const int* __restrict__ idx,
                               const float* __restrict__ mask,
                               const float* __restrict__ deg,
                               const float* __restrict__ bias,
                               float* __restrict__ out, int n)
{
    int g = blockIdx.y;
    int t = blockIdx.x * blockDim.x + threadIdx.x;
    if (t >= n * H) return;
    int i = t / H, f = t % H;
    const float* xwg = xw  + (size_t)g * n * H;
    const float* dgv = deg + (size_t)g * n;
    const int*   ig  = idx + ((size_t)g * n + i) * KNN;
    const float* mg  = mask+ ((size_t)g * n + i) * KNN;
    float di  = dgv[i];
    float acc = xwg[(size_t)i * H + f] / di;
#pragma unroll
    for (int k = 0; k < KNN; ++k) {
        int s   = ig[k];
        float m = mg[k];
        acc += m * rsqrtf(dgv[s] * di) * xwg[(size_t)s * H + f];
    }
    out[(size_t)g * n * H + (size_t)i * H + f] = fmaxf(acc + bias[f], 0.f);
}

// =====================================================================
// Pooling
// =====================================================================
__global__ void pool_feat_kernel(const float* __restrict__ x, float* __restrict__ pooled)
{
    int t = blockIdx.x * blockDim.x + threadIdx.x;
    if (t >= NB * H) return;
    int b = t / H, f = t % H;
    float s = 0.f;
#pragma unroll
    for (int a = 0; a < APB; ++a) s += x[((size_t)b * APB + a) * H + f];
    pooled[t] = s * (1.0f / APB);
}

__global__ void pool_coord_kernel(const float* __restrict__ Z, float* __restrict__ bc)
{
    int t = blockIdx.x * blockDim.x + threadIdx.x;
    if (t >= NB * 3) return;
    int b = t / 3, c = t % 3;
    float s = 0.f;
#pragma unroll
    for (int a = 0; a < APB; ++a) s += Z[((size_t)b * APB + a) * 3 + c];
    bc[t] = s * (1.0f / APB);
}

// =====================================================================
// Edge features into zero-padded 288-column buffer (one block per edge)
// =====================================================================
__global__ void edge_feat_kernel(const float* __restrict__ blocks,
                                 const float* __restrict__ bc,
                                 const int* __restrict__ bidx,
                                 float* __restrict__ ef)
{
    int row = blockIdx.x;              // g*E + (l*KNN + kk)
    int f   = threadIdx.x;             // 0..287
    int g = row / E;
    int e = row % E;
    int l = e / KNN, kk = e % KNN;
    int src = bidx[((size_t)g * L + l) * KNN + kk];
    float* dst = ef + (size_t)row * EFPAD;
    float v = 0.f;
    if (f < H)            v = blocks[((size_t)g * L + src) * H + f];
    else if (f < 2 * H)   v = blocks[((size_t)g * L + l)   * H + (f - H)];
    else if (f == 2 * H) {
        const float* cs = bc + ((size_t)g * L + src) * 3;
        const float* cd = bc + ((size_t)g * L + l)   * 3;
        float dx = cs[0] - cd[0], dy = cs[1] - cd[1], dz = cs[2] - cd[2];
        v = sqrtf(dx * dx + dy * dy + dz * dz + 1e-12f);
    }
    dst[f] = v;
}

// =====================================================================
// Pack Q/K/V into attention-friendly f16:
//   Qh/Kh: [(g*HEADS+h)*E + e][32]   (key/query-major, d contiguous)
//   Vt:    [(g*HEADS+h)*HD + d][E]   (d-major, key contiguous)
// =====================================================================
__global__ void pack_qkv_kernel(const float* __restrict__ Q, const float* __restrict__ K,
                                const float* __restrict__ V,
                                _Float16* __restrict__ Qh, _Float16* __restrict__ Kh,
                                _Float16* __restrict__ Vt)
{
    int t = blockIdx.x * blockDim.x + threadIdx.x;
    if (t >= G * HEADS * E * HD) return;
    int d  = t % HD;
    int e  = (t / HD) % E;
    int gh = t / (HD * E);
    int g  = gh / HEADS, h = gh % HEADS;
    size_t src = ((size_t)g * E + e) * H + h * HD + d;
    Qh[t] = (_Float16)Q[src];
    Kh[t] = (_Float16)K[src];
    Vt[((size_t)gh * HD + d) * E + e] = (_Float16)V[src];
}

// =====================================================================
// Flash attention over the edge set (WMMA f16, f32 accumulate).
// One wave = 16 queries of one (g, head); streams 2048 keys in 32-chunks.
// =====================================================================
__global__ void attn_kernel(const _Float16* __restrict__ Qh, const _Float16* __restrict__ Kh,
                            const _Float16* __restrict__ Vt, const float* __restrict__ emask,
                            float* __restrict__ Om)
{
    __shared__ _Float16 plds[4][16 * 32];
    int wslot = (int)(threadIdx.x >> 5);
    int lane  = (int)(threadIdx.x & 31);
    int wid   = blockIdx.x * 4 + wslot;
    int qt = wid & (E / 16 - 1);           // 0..127
    int h  = (wid >> 7) & (HEADS - 1);
    int g  = wid >> 9;
    int gh = g * HEADS + h;
    int half = lane >> 4, ln = lane & 15;
    _Float16* P = &plds[wslot][0];

    // Q tile, A-layout
    const _Float16* qrow = Qh + ((size_t)gh * E + qt * 16 + ln) * 32;
    v8h q0 = *(const v8h*)(qrow + half * 16);
    v8h q1 = *(const v8h*)(qrow + half * 16 + 8);
    v16h aq;
#pragma unroll
    for (int i = 0; i < 8; ++i) { aq[i] = q0[i]; aq[8 + i] = q1[i]; }

    v8f o0 = {}, o1 = {};
    float mrun[8], lrun[8];
#pragma unroll
    for (int j = 0; j < 8; ++j) { mrun[j] = -1e30f; lrun[j] = 0.f; }

    const _Float16* kbase = Kh + (size_t)gh * E * 32;
    const _Float16* vbase = Vt + (size_t)gh * HD * E;
    const float*    mg    = emask + (size_t)g * E;

    for (int kb = 0; kb < E; kb += 32) {
        // K^T tiles, B-layout: lane = key col (ln), elems = d
        const _Float16* kr0 = kbase + (size_t)(kb + ln) * 32 + half * 16;
        const _Float16* kr1 = kr0 + 16 * 32;
        v8h ka0 = *(const v8h*)(kr0), ka1 = *(const v8h*)(kr0 + 8);
        v8h kb0 = *(const v8h*)(kr1), kb1 = *(const v8h*)(kr1 + 8);
        v16h b0, b1;
#pragma unroll
        for (int i = 0; i < 8; ++i) {
            b0[i] = ka0[i]; b0[8 + i] = ka1[i];
            b1[i] = kb0[i]; b1[8 + i] = kb1[i];
        }
        v8f s0 = {}, s1 = {};
        s0 = __builtin_amdgcn_wmma_f32_16x16x32_f16(false, aq, false, b0, (short)0, s0, false, false);
        s1 = __builtin_amdgcn_wmma_f32_16x16x32_f16(false, aq, false, b1, (short)0, s1, false, false);

        float m0 = mg[kb + ln], m1v = mg[kb + 16 + ln];
#pragma unroll
        for (int j = 0; j < 8; ++j) {
            s0[j] = (m0  > 0.f) ? s0[j] * SCALE : -1e9f;
            s1[j] = (m1v > 0.f) ? s1[j] * SCALE : -1e9f;
        }
        // online softmax: 16-lane-half butterflies
        float cm[8];
#pragma unroll
        for (int j = 0; j < 8; ++j) cm[j] = fmaxf(s0[j], s1[j]);
#pragma unroll
        for (int xm = 1; xm < 16; xm <<= 1)
#pragma unroll
            for (int j = 0; j < 8; ++j) cm[j] = fmaxf(cm[j], __shfl_xor(cm[j], xm, 32));

        float alpha[8];
#pragma unroll
        for (int j = 0; j < 8; ++j) {
            float mn = fmaxf(mrun[j], cm[j]);
            alpha[j] = __expf(mrun[j] - mn);
            mrun[j]  = mn;
        }
        v8f p0, p1; float cs[8];
#pragma unroll
        for (int j = 0; j < 8; ++j) {
            p0[j] = __expf(s0[j] - mrun[j]);
            p1[j] = __expf(s1[j] - mrun[j]);
            cs[j] = p0[j] + p1[j];
        }
#pragma unroll
        for (int xm = 1; xm < 16; xm <<= 1)
#pragma unroll
            for (int j = 0; j < 8; ++j) cs[j] += __shfl_xor(cs[j], xm, 32);
#pragma unroll
        for (int j = 0; j < 8; ++j) {
            lrun[j] = lrun[j] * alpha[j] + cs[j];
            o0[j] *= alpha[j];
            o1[j] *= alpha[j];
        }
        // P relayout: C-layout -> wave-private LDS -> A-layout
#pragma unroll
        for (int j = 0; j < 8; ++j) {
            P[(half * 8 + j) * 32 + ln]      = (_Float16)p0[j];
            P[(half * 8 + j) * 32 + 16 + ln] = (_Float16)p1[j];
        }
        asm volatile("s_wait_dscnt 0" ::: "memory");
        v16h pa;
#pragma unroll
        for (int i = 0; i < 8; ++i) {
            pa[i]     = P[ln * 32 + half * 16 + i];
            pa[8 + i] = P[ln * 32 + half * 16 + 8 + i];
        }
        // V tiles, B-layout: lane = d col, elems = keys (contiguous)
        const _Float16* vr0 = vbase + (size_t)ln * E + kb + half * 16;
        const _Float16* vr1 = vr0 + (size_t)16 * E;
        v8h va0 = *(const v8h*)(vr0), va1 = *(const v8h*)(vr0 + 8);
        v8h vb0 = *(const v8h*)(vr1), vb1 = *(const v8h*)(vr1 + 8);
        v16h bv0, bv1;
#pragma unroll
        for (int i = 0; i < 8; ++i) {
            bv0[i] = va0[i]; bv0[8 + i] = va1[i];
            bv1[i] = vb0[i]; bv1[8 + i] = vb1[i];
        }
        o0 = __builtin_amdgcn_wmma_f32_16x16x32_f16(false, pa, false, bv0, (short)0, o0, false, false);
        o1 = __builtin_amdgcn_wmma_f32_16x16x32_f16(false, pa, false, bv1, (short)0, o1, false, false);
    }
#pragma unroll
    for (int j = 0; j < 8; ++j) {
        float inv = 1.0f / lrun[j];
        int r = qt * 16 + half * 8 + j;
        float* orow = Om + ((size_t)g * E + r) * H + h * HD;
        orow[ln]      = o0[j] * inv;
        orow[16 + ln] = o1[j] * inv;
    }
}

// =====================================================================
// LayerNorm(a + b) per 128-feature row, one wave per row
// =====================================================================
__global__ void ln_res_kernel(const float* __restrict__ a, const float* __restrict__ b,
                              const float* __restrict__ gam, const float* __restrict__ bet,
                              float* __restrict__ out, int rows)
{
    int wave = (int)((blockIdx.x * blockDim.x + threadIdx.x) >> 5);
    int lane = (int)(threadIdx.x & 31);
    if (wave >= rows) return;
    const float* ar = a + (size_t)wave * H;
    const float* br = b + (size_t)wave * H;
    float v[4]; float s = 0.f;
#pragma unroll
    for (int j = 0; j < 4; ++j) { v[j] = ar[lane + 32 * j] + br[lane + 32 * j]; s += v[j]; }
#pragma unroll
    for (int xm = 1; xm < 32; xm <<= 1) s += __shfl_xor(s, xm, 32);
    float mean = s * (1.0f / H);
    float vs = 0.f;
#pragma unroll
    for (int j = 0; j < 4; ++j) { float d = v[j] - mean; vs += d * d; }
#pragma unroll
    for (int xm = 1; xm < 32; xm <<= 1) vs += __shfl_xor(vs, xm, 32);
    float inv = rsqrtf(vs * (1.0f / H) + EPS);
    float* orow = out + (size_t)wave * H;
#pragma unroll
    for (int j = 0; j < 4; ++j) {
        int f = lane + 32 * j;
        orow[f] = (v[j] - mean) * inv * gam[f] + bet[f];
    }
}

// =====================================================================
// scatter-mean edges -> block nodes
// =====================================================================
__global__ void scatter_kernel(const float* __restrict__ e, const float* __restrict__ emask,
                               float* __restrict__ upd)
{
    int t = blockIdx.x * blockDim.x + threadIdx.x;
    if (t >= NB * H) return;
    int b = t / H, f = t % H;
    int g = b / L, l = b % L;
    float s = 0.f, c = 0.f;
#pragma unroll
    for (int kk = 0; kk < KNN; ++kk) {
        size_t row = (size_t)g * E + l * KNN + kk;
        float m = emask[row];
        s += m * e[row * H + f];
        c += m;
    }
    upd[t] = (c > 0.f) ? s / fmaxf(c, 1.0f) : 0.f;
}

// =====================================================================
// Final head (tiny; single block, f32)
// =====================================================================
__global__ void head_kernel(const float* __restrict__ nodes,
                            const float* __restrict__ gpw, const float* __restrict__ gpb,
                            const float* __restrict__ lng, const float* __restrict__ lnb,
                            const float* __restrict__ fw1, const float* __restrict__ fb1,
                            const float* __restrict__ fw2, const float* __restrict__ fb2,
                            const float* __restrict__ fw3, const float* __restrict__ fb3,
                            float* __restrict__ out)
{
    __shared__ float gm[G][H], t1[G][H], t2[G][H];
    __shared__ float red[G * 2];
    int f = threadIdx.x;
    for (int g = 0; g < G; ++g) {
        float s = 0.f;
        for (int l = 0; l < L; ++l) s += nodes[((size_t)g * L + l) * H + f];
        gm[g][f] = s * (1.0f / L);
    }
    __syncthreads();
    for (int g = 0; g < G; ++g) {
        float s = gpb[f];
        for (int k = 0; k < H; ++k) s += gm[g][k] * gpw[k * H + f];
        t1[g][f] = fmaxf(s, 0.f);
    }
    __syncthreads();
    if (f < G) {
        float m = 0.f;
        for (int k = 0; k < H; ++k) m += t1[f][k];
        m *= (1.0f / H);
        float v = 0.f;
        for (int k = 0; k < H; ++k) { float d = t1[f][k] - m; v += d * d; }
        red[f * 2]     = m;
        red[f * 2 + 1] = rsqrtf(v * (1.0f / H) + EPS);
    }
    __syncthreads();
    for (int g = 0; g < G; ++g)
        t2[g][f] = (t1[g][f] - red[g * 2]) * red[g * 2 + 1] * lng[f] + lnb[f];
    __syncthreads();
    for (int g = 0; g < G; ++g) {
        float s = fb1[f];
        for (int k = 0; k < H; ++k) s += t2[g][k] * fw1[k * H + f];
        t1[g][f] = fmaxf(s, 0.f);
    }
    __syncthreads();
    for (int g = 0; g < G; ++g) {
        float s = fb2[f];
        for (int k = 0; k < H; ++k) s += t1[g][k] * fw2[k * H + f];
        t2[g][f] = fmaxf(s, 0.f);
    }
    __syncthreads();
    if (f < G) {
        float s = fb3[0];
        for (int k = 0; k < H; ++k) s += t2[f][k] * fw3[k];
        out[f] = s;
    }
}

// =====================================================================
// host-side launcher
// =====================================================================
static inline void gemm(hipStream_t s, const float* A, int lda, const _Float16* Wt,
                        const float* bias, const float* res, float* C,
                        int M, int Kd, int N, int act)
{
    int waves  = (M / 16) * (N / 64);
    int blocks = (waves + 3) / 4;
    gemm_wmma_kernel<<<blocks, 128, 0, s>>>(A, lda, Wt, bias, res, C, M, Kd, N, act);
}

extern "C" void kernel_launch(void* const* d_in, const int* in_sizes, int n_in,
                              void* d_out, int out_size, void* d_ws, size_t ws_size,
                              hipStream_t stream)
{
    (void)in_sizes; (void)n_in; (void)out_size; (void)ws_size;
    auto F = [&](int i) { return (const float*)d_in[i]; };

    const float* Z   = F(0);
    const int*   A   = (const int*)d_in[1];
    const int*   AP  = (const int*)d_in[2];
    const int*   B   = (const int*)d_in[3];
    const float* atom_emb = F(4);
    const float* pos_emb  = F(5);
    const float* block_emb= F(6);
    const float* gcn0w = F(7),  *gcn0b = F(8);
    const float* gcn1w = F(9),  *gcn1b = F(10);
    const float* pw = F(11), *pb = F(12);
    const float* ew1 = F(13), *eb1 = F(14);
    const float* ew2 = F(15), *eb2 = F(16);
    const float* gpw = F(53), *gpb = F(54);
    const float* lng = F(55), *lnb = F(56);
    const float* fw1 = F(57), *fb1 = F(58);
    const float* fw2 = F(59), *fb2 = F(60);
    const float* fw3 = F(61), *fb3 = F(62);

    // ---- workspace layout ----
    char* wp = (char*)d_ws;
    auto allocb = [&](size_t bytes) { void* p = (void*)wp; wp += ((bytes + 255) & ~size_t(255)); return p; };
    auto allocf = [&](size_t n) { return (float*)allocb(n * 4); };
    auto alloch = [&](size_t n) { return (_Float16*)allocb(n * 2); };

    float* x     = allocf((size_t)NA * H);
    float* xw    = allocf((size_t)NA * H);
    float* x2    = allocf((size_t)NA * H);
    int*   aidx  = (int*)allocb((size_t)NA * KNN * 4);
    float* amask = allocf((size_t)NA * KNN);
    float* adeg  = allocf(NA);
    float* pooled= allocf((size_t)NB * H);
    float* nodesA= allocf((size_t)NB * H);
    float* nodesB= allocf((size_t)NB * H);
    float* upd   = allocf((size_t)NB * H);
    float* bc    = allocf((size_t)NB * 3);
    int*   bidx  = (int*)allocb((size_t)NB * KNN * 4);
    float* bmask = allocf((size_t)NB * KNN);      // == emask [G,E]
    float* bdeg  = allocf(NB);
    float* ef    = allocf((size_t)G * E * EFPAD);
    float* t1    = allocf((size_t)G * E * H);
    float* t2    = allocf((size_t)G * E * H);
    float* t3    = allocf((size_t)G * E * H);
    float* t4    = allocf((size_t)G * E * H);
    float* t5    = allocf((size_t)G * E * H);
    _Float16* Qh = alloch((size_t)G * HEADS * E * HD);
    _Float16* Kh = alloch((size_t)G * HEADS * E * HD);
    _Float16* Vt = alloch((size_t)G * HEADS * E * HD);

    // ---- pack all WMMA weights to transposed f16 ----
    auto packw = [&](const float* W, int Kd, int Kpad) {
        _Float16* p = alloch((size_t)H * Kpad);
        pack_w_kernel<<<(H * Kpad + 255) / 256, 256, 0, stream>>>(W, p, Kd, Kpad, H);
        return p;
    };
    _Float16* gcn0wh = packw(gcn0w, H, H);
    _Float16* gcn1wh = packw(gcn1w, H, H);
    _Float16* pwh    = packw(pw,    H, H);
    _Float16* ew1h   = packw(ew1,   2 * H + 1, EFPAD);
    _Float16* ew2h   = packw(ew2,   H, H);
    _Float16* esaW[2][7];
    for (int layer = 0; layer < 2; ++layer) {
        int base = 17 + 18 * layer;
        static const int widx[7] = {0, 2, 4, 6, 8, 10, 12}; // wq wk wv wo mw1 mw2 nw
        for (int wti = 0; wti < 7; ++wti)
            esaW[layer][wti] = packw(F(base + widx[wti]), H, H);
    }

    // 1. embeddings
    embed_kernel<<<(NA * H) / 256, 256, 0, stream>>>(A, AP, B, atom_emb, pos_emb, block_emb, x);
    // 2. atom-level radius graph
    knn_kernel<<<dim3((NPG + 63) / 64, G), 64, 0, stream>>>(Z, aidx, amask, adeg, NPG, 36.0f);
    // 3. GCN x2
    gemm(stream, x, H, gcn0wh, nullptr, nullptr, xw, NA, H, H, ACT_NONE);
    gcn_agg_kernel<<<dim3((NPG * H + 255) / 256, G), 256, 0, stream>>>(xw, aidx, amask, adeg, gcn0b, x2, NPG);
    gemm(stream, x2, H, gcn1wh, nullptr, nullptr, xw, NA, H, H, ACT_NONE);
    gcn_agg_kernel<<<dim3((NPG * H + 255) / 256, G), 256, 0, stream>>>(xw, aidx, amask, adeg, gcn1b, x, NPG);
    // 4. pool atoms -> blocks
    pool_feat_kernel<<<(NB * H) / 256, 256, 0, stream>>>(x, pooled);
    pool_coord_kernel<<<(NB * 3 + 255) / 256, 256, 0, stream>>>(Z, bc);
    gemm(stream, pooled, H, pwh, pb, nullptr, nodesA, NB, H, H, ACT_RELU);
    // 5. block-level radius graph (r = 9 -> r2 = 81)
    knn_kernel<<<dim3((L + 63) / 64, G), 64, 0, stream>>>(bc, bidx, bmask, bdeg, L, 81.0f);
    // 6. edge features + edge MLP
    edge_feat_kernel<<<G * E, EFPAD, 0, stream>>>(nodesA, bc, bidx, ef);
    gemm(stream, ef, EFPAD, ew1h, eb1, nullptr, t1, G * E, EFPAD, H, ACT_RELU);
    gemm(stream, t1, H, ew2h, eb2, nullptr, t2, G * E, H, H, ACT_NONE);   // e lives in t2
    // 7. ESA x2
    float* nodesIn = nodesA;
    float* nodesOut = nodesB;
    for (int layer = 0; layer < 2; ++layer) {
        int base = 17 + 18 * layer;
        const float* bq = F(base + 1),  *bk = F(base + 3),  *bv = F(base + 5);
        const float* bo = F(base + 7),  *mb1= F(base + 9),  *mb2= F(base + 11);
        const float* nb = F(base + 13);
        const float* l1g= F(base + 14), *l1b= F(base + 15);
        const float* l2g= F(base + 16), *l2b= F(base + 17);
        _Float16 *wqh = esaW[layer][0], *wkh = esaW[layer][1], *wvh = esaW[layer][2];
        _Float16 *woh = esaW[layer][3], *mw1h = esaW[layer][4], *mw2h = esaW[layer][5];
        _Float16 *nwh = esaW[layer][6];

        gemm(stream, t2, H, wqh, bq, nullptr, t1, G * E, H, H, ACT_NONE);   // Q
        gemm(stream, t2, H, wkh, bk, nullptr, t3, G * E, H, H, ACT_NONE);   // K
        gemm(stream, t2, H, wvh, bv, nullptr, t4, G * E, H, H, ACT_NONE);   // V
        pack_qkv_kernel<<<(G * HEADS * E * HD) / 256, 256, 0, stream>>>(t1, t3, t4, Qh, Kh, Vt);
        attn_kernel<<<(G * HEADS * (E / 16)) / 4, 128, 0, stream>>>(Qh, Kh, Vt, bmask, t5);
        gemm(stream, t5, H, woh, bo, nullptr, t1, G * E, H, H, ACT_NONE);   // out-proj
        ln_res_kernel<<<(G * E) / 8, 256, 0, stream>>>(t2, t1, l1g, l1b, t3, G * E);
        gemm(stream, t3, H, mw1h, mb1, nullptr, t4, G * E, H, H, ACT_RELU);
        gemm(stream, t4, H, mw2h, mb2, nullptr, t5, G * E, H, H, ACT_NONE);
        ln_res_kernel<<<(G * E) / 8, 256, 0, stream>>>(t3, t5, l2g, l2b, t2, G * E);
        scatter_kernel<<<(NB * H) / 256, 256, 0, stream>>>(t2, bmask, upd);
        gemm(stream, upd, H, nwh, nb, nodesIn, nodesOut, NB, H, H, ACT_RELU);
        float* tmp = nodesIn; nodesIn = nodesOut; nodesOut = tmp;
    }
    // 8. head
    head_kernel<<<1, 128, 0, stream>>>(nodesIn, gpw, gpb, lng, lnb,
                                       fw1, fb1, fw2, fb2, fw3, fb3, (float*)d_out);
}
